// WindowTransformerBlock_27135603376351
// MI455X (gfx1250) — matrix operand order repaired and checked by reading, not compiled, not run on hardware
//
#include <hip/hip_runtime.h>

// ---------- types ----------
typedef __attribute__((ext_vector_type(16))) __bf16 v16bf;
typedef __attribute__((ext_vector_type(8)))  float  v8f;
typedef __attribute__((ext_vector_type(4)))  unsigned int u32x4;
typedef __attribute__((ext_vector_type(4)))  float  f32x4;

union FragBF { v16bf v; u32x4 q[2]; };
union Pack8  { u32x4 q; __bf16 h[8]; };

#define T_TOK   262144   // 4 * 256 * 256 tokens
#define CDIM    256

__device__ __forceinline__ float wave_sum32(float v) {
#pragma unroll
  for (int m = 1; m < 32; m <<= 1) v += __shfl_xor(v, m, 32);
  return v;
}

// async DMA: 16B from global (per-lane addr) into LDS (per-lane offset); ASYNCcnt
__device__ __forceinline__ void async_b128(const void* gptr, void* lptr) {
  unsigned lds = (unsigned)(uintptr_t)lptr;   // low 32 bits of flat = LDS offset
  asm volatile("global_load_async_to_lds_b128 %0, %1, off"
               :: "v"(lds), "v"(gptr) : "memory");
}
__device__ __forceinline__ void wait_async0() {
  asm volatile("s_wait_asynccnt 0x0" ::: "memory");
}

// ---------- weight convert + transpose: W[K][N] f32 -> WT[N][K] bf16 ----------
__global__ __launch_bounds__(256)
void convert_wt(const float* __restrict__ W, __bf16* __restrict__ WT, int K, int N) {
  long idx = (long)blockIdx.x * 256 + threadIdx.x;
  if (idx < (long)K * N) {
    int k = (int)(idx / N), n = (int)(idx % N);
    WT[(long)n * K + k] = (__bf16)W[idx];
  }
}

// ---------- layernorm: f32 [T,256] -> bf16 [T,256]; one wave per token ----------
__global__ __launch_bounds__(256)
void ln_kernel(const float* __restrict__ x, const float* __restrict__ g,
               const float* __restrict__ b, __bf16* __restrict__ out) {
  const int wave = threadIdx.x >> 5, lane = threadIdx.x & 31;
  const long row = (long)blockIdx.x * 8 + wave;
  const float* xr = x + row * CDIM + lane * 8;
  f32x4 a0 = *(const f32x4*)xr;
  f32x4 a1 = *(const f32x4*)(xr + 4);
  float v[8] = {a0.x, a0.y, a0.z, a0.w, a1.x, a1.y, a1.z, a1.w};
  float s = 0.f;
#pragma unroll
  for (int i = 0; i < 8; ++i) s += v[i];
  s = wave_sum32(s);
  float mu = s * (1.f / 256.f);
  float q = 0.f;
#pragma unroll
  for (int i = 0; i < 8; ++i) { float d = v[i] - mu; q += d * d; }
  q = wave_sum32(q);
  float rs = rsqrtf(q * (1.f / 256.f) + 1e-5f);
  Pack8 p;
#pragma unroll
  for (int i = 0; i < 8; ++i) {
    float gg = g[lane * 8 + i], bb = b[lane * 8 + i];
    p.h[i] = (__bf16)((v[i] - mu) * rs * gg + bb);
  }
  *(u32x4*)(out + row * CDIM + lane * 8) = p.q;
}

// ---------- tiled bf16 WMMA GEMM, double-buffered async-to-LDS staging ----------
// EPI 0: store bf16   EPI 1: gelu(exact erf) -> bf16   EPI 2: + resid -> f32
template <int EPI>
__global__ __launch_bounds__(256)
void gemm_bf16(const __bf16* __restrict__ A, const __bf16* __restrict__ WT,
               const float* __restrict__ bias, const float* __restrict__ resid,
               void* __restrict__ outp, int M, int N, int K) {
  __shared__ __bf16 As[2][128][40];   // +8 bf16 pad (row stride 80B, 16B aligned)
  __shared__ __bf16 Bs[2][128][40];
  const int tid = threadIdx.x;
  const int nTiles = N >> 7;
  const int bm = blockIdx.x / nTiles;
  const int bn = blockIdx.x % nTiles;
  const int wave = tid >> 5, lane = tid & 31;
  const int half = lane >> 4, l16 = lane & 15;
  const int waveM = wave >> 2, waveN = wave & 3;

  // issue async copies of K-tile `ks` into buffer `bufi` (no VGPR data regs)
  auto issue = [&](int bufi, int ks) {
    const int k0 = ks << 5;
#pragma unroll
    for (int i = 0; i < 2; ++i) {
      int idx = tid + (i << 8);            // 0..511
      int row = idx >> 2, c = (idx & 3) << 3;
      async_b128(A  + (long)(bm * 128 + row) * K + k0 + c, &As[bufi][row][c]);
      async_b128(WT + (long)(bn * 128 + row) * K + k0 + c, &Bs[bufi][row][c]);
    }
  };

  v8f acc[4][2];
#pragma unroll
  for (int i = 0; i < 4; ++i)
#pragma unroll
    for (int j = 0; j < 2; ++j)
      acc[i][j] = (v8f){0.f, 0.f, 0.f, 0.f, 0.f, 0.f, 0.f, 0.f};

  const int kSteps = K >> 5;
  issue(0, 0);
  int cur = 0;
  for (int ks = 0; ks < kSteps; ++ks) {
    wait_async0();          // this wave's copies into `cur` have landed
    __syncthreads();        // every wave's copies have landed
    if (ks + 1 < kSteps) issue(cur ^ 1, ks + 1);   // prefetch next tile

    FragBF af[4], bf[2];
#pragma unroll
    for (int mt = 0; mt < 4; ++mt) {       // A frag: 16x32, lane=M, k=half*8(+16)
      int row = waveM * 64 + mt * 16 + l16;
      af[mt].q[0] = *(const u32x4*)&As[cur][row][half * 8];
      af[mt].q[1] = *(const u32x4*)&As[cur][row][16 + half * 8];
    }
#pragma unroll
    for (int nt = 0; nt < 2; ++nt) {       // B frag: lane=N, k=half*8(+16)
      int row = waveN * 32 + nt * 16 + l16;
      bf[nt].q[0] = *(const u32x4*)&Bs[cur][row][half * 8];
      bf[nt].q[1] = *(const u32x4*)&Bs[cur][row][16 + half * 8];
    }
#pragma unroll
    for (int mt = 0; mt < 4; ++mt)
#pragma unroll
      for (int nt = 0; nt < 2; ++nt)
        acc[mt][nt] = __builtin_amdgcn_wmma_f32_16x16x32_bf16(
            false, af[mt].v, false, bf[nt].v, (short)0, acc[mt][nt], false, false);
    cur ^= 1;
  }

  // epilogue
#pragma unroll
  for (int nt = 0; nt < 2; ++nt) {
    int col = bn * 128 + waveN * 32 + nt * 16 + l16;
    float bv = bias[col];
#pragma unroll
    for (int mt = 0; mt < 4; ++mt) {
#pragma unroll
      for (int r = 0; r < 8; ++r) {
        long row = (long)bm * 128 + waveM * 64 + mt * 16 + half * 8 + r;
        long o = row * N + col;
        float v = acc[mt][nt][r] + bv;
        if (EPI == 0) {
          ((__bf16*)outp)[o] = (__bf16)v;
        } else if (EPI == 1) {
          v = 0.5f * v * (1.0f + erff(v * 0.70710678118654752f));
          ((__bf16*)outp)[o] = (__bf16)v;
        } else {
          ((float*)outp)[o] = v + resid[o];
        }
      }
    }
  }
}

// ---------- window attention: one block per (window, head-pair) ----------
// qkv: bf16 [T,768] laid out per token as [3][NH][HD]; out: bf16 [T,256]
__global__ __launch_bounds__(256)
void attn_kernel(const __bf16* __restrict__ qkv, const float* __restrict__ bias_table,
                 __bf16* __restrict__ out) {
  __shared__ __bf16 qloc[2][64][32];
  __shared__ __bf16 kloc[2][64][32];
  __shared__ __bf16 vT[2][32][64];       // V transposed: [head][d][token]
  __shared__ __bf16 pmat[8][16][64];     // per-wave softmax probs
  __shared__ float  biasLds[225 * 8];

  const int tid = threadIdx.x;
  const int blk = blockIdx.x;
  const int wid = blk >> 2;              // window id 0..4095
  const int hp  = blk & 3;               // head pair 0..3
  const int b   = wid >> 10;
  const int rem = wid & 1023;
  const int wy  = rem >> 5;
  const int wx  = rem & 31;
  const long rowBase = (long)b * 65536 + (long)wy * 8 * 256 + wx * 8;

  for (int i = tid; i < 225 * 8; i += 256) biasLds[i] = bias_table[i];

  // stage Q/K for the 2 heads via async DMA: 1024 x 16B
#pragma unroll
  for (int i = 0; i < 4; ++i) {
    int idx = tid + i * 256;
    int t = idx >> 4, r = idx & 15;
    int hh = r >> 3, qk = (r >> 2) & 1, c = r & 3;
    long grow = rowBase + (t >> 3) * 256 + (t & 7);
    const __bf16* src = qkv + grow * 768 + qk * 256 + (hp * 2 + hh) * 32 + c * 8;
    if (qk) async_b128(src, &kloc[hh][t][c * 8]);
    else    async_b128(src, &qloc[hh][t][c * 8]);
  }
  // stage V transposed (needs VGPR transpose): 512 x 16B reads, 8 x b16 scatter
#pragma unroll
  for (int i = 0; i < 2; ++i) {
    int idx = tid + i * 256;
    int t = idx >> 3, r = idx & 7;
    int hh = r >> 2, c = r & 3;
    long grow = rowBase + (t >> 3) * 256 + (t & 7);
    Pack8 d;
    d.q = *(const u32x4*)(qkv + grow * 768 + 512 + (hp * 2 + hh) * 32 + c * 8);
#pragma unroll
    for (int j = 0; j < 8; ++j) vT[hh][c * 8 + j][t] = d.h[j];
  }
  wait_async0();
  __syncthreads();

  const int wave = tid >> 5, lane = tid & 31;
  const int half = lane >> 4, l16 = lane & 15;
  const int hh = wave >> 2;              // head within pair
  const int mt = wave & 3;               // 16-row tile of the window
  const int h  = hp * 2 + hh;

  // scores = Q @ K^T  (K rows load directly as B fragments: k-dim = HD)
  FragBF qf;
  {
    int row = mt * 16 + l16;
    qf.q[0] = *(const u32x4*)&qloc[hh][row][half * 8];
    qf.q[1] = *(const u32x4*)&qloc[hh][row][16 + half * 8];
  }
  v8f sc[4];
#pragma unroll
  for (int nt = 0; nt < 4; ++nt) {
    FragBF kf;
    int row = nt * 16 + l16;
    kf.q[0] = *(const u32x4*)&kloc[hh][row][half * 8];
    kf.q[1] = *(const u32x4*)&kloc[hh][row][16 + half * 8];
    v8f z = (v8f){0.f, 0.f, 0.f, 0.f, 0.f, 0.f, 0.f, 0.f};
    sc[nt] = __builtin_amdgcn_wmma_f32_16x16x32_bf16(
        false, qf.v, false, kf.v, (short)0, z, false, false);
  }

  // scale + relative-position bias + row softmax (rows live on lane halves)
  const float scale = 0.17677669529663687f;  // 1/sqrt(32)
#pragma unroll
  for (int r = 0; r < 8; ++r) {
    int tm = mt * 16 + half * 8 + r;
    int yq = tm >> 3, xq = tm & 7;
    float vals[4];
    float mx = -1e30f;
#pragma unroll
    for (int nt = 0; nt < 4; ++nt) {
      int tn = nt * 16 + l16;
      int yk = tn >> 3, xk = tn & 7;
      int bidx = (yq - yk + 7) * 15 + (xq - xk + 7);
      float v = sc[nt][r] * scale + biasLds[bidx * 8 + h];
      vals[nt] = v;
      mx = fmaxf(mx, v);
    }
#pragma unroll
    for (int m = 1; m < 16; m <<= 1) mx = fmaxf(mx, __shfl_xor(mx, m, 32));
    float sum = 0.f;
#pragma unroll
    for (int nt = 0; nt < 4; ++nt) { vals[nt] = __expf(vals[nt] - mx); sum += vals[nt]; }
#pragma unroll
    for (int m = 1; m < 16; m <<= 1) sum += __shfl_xor(sum, m, 32);
    float inv = 1.f / sum;
#pragma unroll
    for (int nt = 0; nt < 4; ++nt)
      pmat[wave][half * 8 + r][nt * 16 + l16] = (__bf16)(vals[nt] * inv);
  }
  // same-wave DS writes then reads are in-order; no workgroup barrier needed

  // out = P @ V  (V^T rows load as B fragments)
  v8f o[2] = {(v8f){0.f,0.f,0.f,0.f,0.f,0.f,0.f,0.f},
              (v8f){0.f,0.f,0.f,0.f,0.f,0.f,0.f,0.f}};
#pragma unroll
  for (int kk = 0; kk < 2; ++kk) {
    FragBF pf;
    pf.q[0] = *(const u32x4*)&pmat[wave][l16][kk * 32 + half * 8];
    pf.q[1] = *(const u32x4*)&pmat[wave][l16][kk * 32 + 16 + half * 8];
#pragma unroll
    for (int ntd = 0; ntd < 2; ++ntd) {
      FragBF vf;
      int drow = ntd * 16 + l16;
      vf.q[0] = *(const u32x4*)&vT[hh][drow][kk * 32 + half * 8];
      vf.q[1] = *(const u32x4*)&vT[hh][drow][kk * 32 + 16 + half * 8];
      o[ntd] = __builtin_amdgcn_wmma_f32_16x16x32_bf16(
          false, pf.v, false, vf.v, (short)0, o[ntd], false, false);
    }
  }

  // scatter to attn_out bf16 [T,256]
#pragma unroll
  for (int ntd = 0; ntd < 2; ++ntd) {
#pragma unroll
    for (int r = 0; r < 8; ++r) {
      int tm = mt * 16 + half * 8 + r;
      long grow = rowBase + (tm >> 3) * 256 + (tm & 7);
      out[grow * 256 + h * 32 + ntd * 16 + l16] = (__bf16)o[ntd][r];
    }
  }
}

// ---------- launcher ----------
extern "C" void kernel_launch(void* const* d_in, const int* in_sizes, int n_in,
                              void* d_out, int out_size, void* d_ws, size_t ws_size,
                              hipStream_t stream) {
  const float* x       = (const float*)d_in[0];
  const float* ln1_g   = (const float*)d_in[1];
  const float* ln1_b   = (const float*)d_in[2];
  const float* qkv_w   = (const float*)d_in[3];
  const float* qkv_b   = (const float*)d_in[4];
  const float* proj_w  = (const float*)d_in[5];
  const float* proj_b  = (const float*)d_in[6];
  const float* btab    = (const float*)d_in[7];
  const float* ln2_g   = (const float*)d_in[8];
  const float* ln2_b   = (const float*)d_in[9];
  const float* fc1_w   = (const float*)d_in[10];
  const float* fc1_b   = (const float*)d_in[11];
  const float* fc2_w   = (const float*)d_in[12];
  const float* fc2_b   = (const float*)d_in[13];

  char* ws = (char*)d_ws;
  __bf16* wqkvT  = (__bf16*)(ws + 0);                        // 768x256 bf16
  __bf16* wprojT = (__bf16*)(ws + 393216);                   // 256x256
  __bf16* wfc1T  = (__bf16*)(ws + 524288);                   // 1024x256
  __bf16* wfc2T  = (__bf16*)(ws + 1048576);                  // 256x1024
  __bf16* bufA   = (__bf16*)(ws + 1572864);                  // T x 1024 bf16 (qkv / fc1 act)
  __bf16* bufB   = (__bf16*)(ws + 1572864 + 536870912ULL);   // T x 256 bf16 (reused)

  const int T = T_TOK;

  convert_wt<<<(256 * 768 + 255) / 256, 256, 0, stream>>>(qkv_w,  wqkvT,  256, 768);
  convert_wt<<<(256 * 256 + 255) / 256, 256, 0, stream>>>(proj_w, wprojT, 256, 256);
  convert_wt<<<(256 * 1024 + 255) / 256, 256, 0, stream>>>(fc1_w, wfc1T, 256, 1024);
  convert_wt<<<(1024 * 256 + 255) / 256, 256, 0, stream>>>(fc2_w, wfc2T, 1024, 256);

  // h = LN1(x) -> bf16
  ln_kernel<<<T / 8, 256, 0, stream>>>(x, ln1_g, ln1_b, bufB);
  // qkv = h @ Wqkv + b -> bf16 [T,768]
  gemm_bf16<0><<<(T / 128) * (768 / 128), 256, 0, stream>>>(
      bufB, wqkvT, qkv_b, nullptr, bufA, T, 768, 256);
  // window attention -> bf16 [T,256]
  attn_kernel<<<4096 * 4, 256, 0, stream>>>(bufA, btab, bufB);
  // h2 = attn @ Wproj + b + x -> f32 in d_out
  gemm_bf16<2><<<(T / 128) * (256 / 128), 256, 0, stream>>>(
      bufB, wprojT, proj_b, x, d_out, T, 256, 256);
  // LN2(h2) -> bf16
  ln_kernel<<<T / 8, 256, 0, stream>>>((const float*)d_out, ln2_g, ln2_b, bufB);
  // gelu(LN2 @ Wfc1 + b) -> bf16 [T,1024]
  gemm_bf16<1><<<(T / 128) * (1024 / 128), 256, 0, stream>>>(
      bufB, wfc1T, fc1_b, nullptr, bufA, T, 1024, 256);
  // out = h2 + (act @ Wfc2 + b)  (in-place residual from d_out)
  gemm_bf16<2><<<(T / 128) * (256 / 128), 256, 0, stream>>>(
      bufA, wfc2T, fc2_b, (const float*)d_out, d_out, T, 256, 1024);
}
